// HierarchicalCLIPLoss_34428457845256
// MI455X (gfx1250) — matrix operand order repaired and checked by reading, not compiled, not run on hardware
//
#include <hip/hip_runtime.h>
#include <hip/hip_bf16.h>

typedef __attribute__((ext_vector_type(16))) __bf16 v16bf;
typedef __attribute__((ext_vector_type(8)))  float  v8f;

#define BSZ   16
#define SDIM  4096
#define TDIM  64
#define EDIM  128
#define CHUNK 128
#define NCHUNK (SDIM / CHUNK)
#define VST 136   // vn_lds row stride in bf16 elems (128 + 8 pad -> 272B, bank-spread)
#define PST 136   // P row stride in bf16 elems

static __device__ __forceinline__ void wait_async0() {
#if defined(__has_builtin)
#if __has_builtin(__builtin_amdgcn_s_wait_asynccnt)
  __builtin_amdgcn_s_wait_asynccnt(0);
  return;
#endif
#endif
  asm volatile("s_wait_asynccnt 0" ::: "memory");
}

// ---------------------------------------------------------------------------
// Kernel 0: L2-normalize rows of vis (-> bf16) and txt (-> bf16 + f32 + w1)
// One wave per 128-elem row.
// ---------------------------------------------------------------------------
__global__ void norm_kernel(const float* __restrict__ vis, const float* __restrict__ txt,
                            __bf16* __restrict__ vnb, __bf16* __restrict__ tnb,
                            float* __restrict__ tnf, float* __restrict__ w1) {
  const int wv = threadIdx.x >> 5, lane = threadIdx.x & 31;
  const long long NV = (long long)BSZ * SDIM;   // 65536
  const long long NT = (long long)BSZ * TDIM;   // 1024
  long long row = (long long)blockIdx.x * 8 + wv;
  if (row >= NV + NT) return;
  const bool isTxt = row >= NV;
  const float* src = isTxt ? (txt + (row - NV) * EDIM) : (vis + row * EDIM);
  float4 x = *reinterpret_cast<const float4*>(src + lane * 4);
  float ss = x.x * x.x + x.y * x.y + x.z * x.z + x.w * x.w;
  for (int m = 16; m; m >>= 1) ss += __shfl_xor(ss, m);
  float nrm = sqrtf(ss);
  float scale = 1.0f / fmaxf(nrm, 1e-12f);   // F.normalize eps
  float4 y; y.x = x.x * scale; y.y = x.y * scale; y.z = x.z * scale; y.w = x.w * scale;
  __bf16 o[4] = { (__bf16)y.x, (__bf16)y.y, (__bf16)y.z, (__bf16)y.w };
  if (!isTxt) {
    *reinterpret_cast<uint2*>(vnb + row * EDIM + lane * 4) = *reinterpret_cast<uint2*>(o);
  } else {
    long long rt = row - NV;
    *reinterpret_cast<uint2*>(tnb + rt * EDIM + lane * 4) = *reinterpret_cast<uint2*>(o);
    *reinterpret_cast<float4*>(tnf + rt * EDIM + lane * 4) = y;
    if (lane == 0) w1[rt] = nrm * scale;     // ||tn|| after normalize (~1)
  }
}

// ---------------------------------------------------------------------------
// Kernel 1: fused attention pipeline, one workgroup per (i_txt, b_vis) pair.
// ---------------------------------------------------------------------------
__global__ __launch_bounds__(256)
void pair_kernel(const __bf16* __restrict__ vnb, const __bf16* __restrict__ tnb,
                 const float* __restrict__ tnf, const float* __restrict__ w1g,
                 float* __restrict__ simbuf, float* __restrict__ lglob,
                 float* __restrict__ att /* = d_out + 1 */) {
  __shared__ __align__(16) __bf16 vn_lds[CHUNK * VST];  // 34816 B
  __shared__ __align__(16) __bf16 P[TDIM * PST];        // 17408 B, p stored t-major
  __shared__ float l_lds[TDIM], num_lds[TDIM], sq_lds[TDIM], red_lds[TDIM];

  const int tid  = threadIdx.x;
  const int wv   = tid >> 5;        // wave id 0..7
  const int lane = tid & 31;
  const int n    = lane & 15;       // N column within a 16x16 tile
  const int h    = lane >> 4;       // lane half
  const int b    = blockIdx.x & 15; // vis sample
  const int i    = blockIdx.x >> 4; // txt sample
  const bool diag = (i == b);

  if (tid < TDIM) { l_lds[tid] = 0.f; num_lds[tid] = 0.f; sq_lds[tid] = 0.f; }

  const __bf16* vnB = vnb + (long long)b * SDIM * EDIM;
  const __bf16* tnI = tnb + (long long)i * TDIM * EDIM;

  // per-thread cooperative-load addressing (2 threads per 256B row)
  const int ld_r    = tid >> 1;
  const int ld_part = tid & 1;
  const unsigned lds_base =
      (unsigned)(size_t)&vn_lds[ld_r * VST + ld_part * 64];   // low 32 bits = LDS offset

  // Phase-B accumulators: acc[t = 16*mt + 8*h + v][e = 16*wv + n]
  v8f acc2[4];
  #pragma unroll
  for (int mt = 0; mt < 4; ++mt) { v8f zz = {}; acc2[mt] = zz; }

  __syncthreads();

  for (int ch = 0; ch < NCHUNK; ++ch) {
    const int s0 = ch * CHUNK;

    // ---- async cooperative load: vn chunk (128 x 128 bf16) -> LDS, no VGPR round-trip
    // offset immediate applies to BOTH lds and global addresses (ISA 08 sec 4.4)
    {
      unsigned goff = (unsigned)((s0 + ld_r) * (EDIM * 2) + ld_part * 128);
#define ALOAD(OFF) asm volatile(                                             \
      "global_load_async_to_lds_b128 %0, %1, %2 offset:" #OFF               \
      :: "v"(lds_base), "v"(goff), "s"(vnB) : "memory")
      ALOAD(0);  ALOAD(16); ALOAD(32); ALOAD(48);
      ALOAD(64); ALOAD(80); ALOAD(96); ALOAD(112);
#undef ALOAD
      wait_async0();
    }
    __syncthreads();

    // ---- Phase A: z(16x64) WMMA, softmax over t, p = exp(4*s1-4), l, P ----
    {
      v8f zc[4];
      #pragma unroll
      for (int t4 = 0; t4 < 4; ++t4) { v8f zz = {}; zc[t4] = zz; }

      #pragma unroll
      for (int kb = 0; kb < 4; ++kb) {
        // A fragment (16-bit A 16x32 layout): lane row m=n, K runs [8h..]+[16+8h..]
        union { v16bf v; uint4 q[2]; } ua;
        {
          const __bf16* p = &vn_lds[(16 * wv + n) * VST + 32 * kb];
          ua.q[0] = *reinterpret_cast<const uint4*>(p + 8 * h);
          ua.q[1] = *reinterpret_cast<const uint4*>(p + 16 + 8 * h);
        }
        #pragma unroll
        for (int t4 = 0; t4 < 4; ++t4) {
          // B fragment: column N = t = 16*t4 + n; K = e runs [8h..]+[16+8h..] in tn row
          union { v16bf v; uint4 q[2]; } ub;
          const __bf16* p = tnI + (16 * t4 + n) * EDIM + 32 * kb;
          ub.q[0] = *reinterpret_cast<const uint4*>(p + 8 * h);
          ub.q[1] = *reinterpret_cast<const uint4*>(p + 16 + 8 * h);
          zc[t4] = __builtin_amdgcn_wmma_f32_16x16x32_bf16(
              false, ua.v, false, ub.v, (short)0, zc[t4], false, false);
        }
      }

      // softmax over t: z in [-1,1] -> no max needed
      float ez[4][8];
      float rs[8];
      #pragma unroll
      for (int v = 0; v < 8; ++v) rs[v] = 0.f;
      #pragma unroll
      for (int t4 = 0; t4 < 4; ++t4)
        #pragma unroll
        for (int v = 0; v < 8; ++v) { float e = __expf(zc[t4][v]); ez[t4][v] = e; rs[v] += e; }
      #pragma unroll
      for (int v = 0; v < 8; ++v) {   // sum over 16 n-lanes (stays within half)
        float r = rs[v];
        r += __shfl_xor(r, 1); r += __shfl_xor(r, 2);
        r += __shfl_xor(r, 4); r += __shfl_xor(r, 8);
        rs[v] = 4.0f * __builtin_amdgcn_rcpf(r);   // fold TEMP1; v_rcp_f32, no slow div
      }
      // p = exp(TEMP1*s1 - 4)  (constant max = 4 since 4*s1 <= 4, exact softmax)
      #pragma unroll
      for (int t4 = 0; t4 < 4; ++t4) {
        float colp = 0.f;
        #pragma unroll
        for (int v = 0; v < 8; ++v) {
          float pv = __expf(__builtin_fmaf(ez[t4][v], rs[v], -4.0f));
          ez[t4][v] = pv;
          colp += pv;
          P[(16 * t4 + n) * PST + 16 * wv + 8 * h + v] = (__bf16)pv;
        }
        colp += __shfl_xor(colp, 16);          // combine row-halves
        if (h == 0) atomicAdd(&l_lds[16 * t4 + n], colp);
      }
      if (diag) {                               // raw p -> att_maps (scaled later)
        #pragma unroll
        for (int t4 = 0; t4 < 4; ++t4) {
          int t = 16 * t4 + n;
          float* dst = att + (long long)(i * TDIM + t) * SDIM + s0 + 16 * wv + 8 * h;
          #pragma unroll
          for (int v = 0; v < 8; ++v) dst[v] = ez[t4][v];
        }
      }
    }
    __syncthreads();

    // ---- Phase B: acc2 += P (64 x 128s) * vn (128s x Eslice), per-wave E slice ----
    {
      const int ecol = 16 * wv + n;
      #pragma unroll
      for (int ks = 0; ks < 4; ++ks) {
        v16bf bf;                                // B[k=s][n=e] column gather
        #pragma unroll
        for (int j = 0; j < 16; ++j)
          bf[j] = vn_lds[(32 * ks + 16 * h + j) * VST + ecol];
        #pragma unroll
        for (int mt = 0; mt < 4; ++mt) {
          union { v16bf v; uint4 q[2]; } ua;    // A[m=t][k=s] from t-major P
          const __bf16* p = &P[(16 * mt + n) * PST + 32 * ks];
          ua.q[0] = *reinterpret_cast<const uint4*>(p + 16 * h);
          ua.q[1] = *reinterpret_cast<const uint4*>(p + 16 * h + 8);
          acc2[mt] = __builtin_amdgcn_wmma_f32_16x16x32_bf16(
              false, ua.v, false, bf, (short)0, acc2[mt], false, false);
        }
      }
    }
    __syncthreads();
  }

  // ---- finalize: num, ||wc||, cos, logsumexp over t ----
  {
    const int ecol = 16 * wv + n;
    #pragma unroll
    for (int mt = 0; mt < 4; ++mt)
      #pragma unroll
      for (int v = 0; v < 8; ++v) {
        int t = 16 * mt + 8 * h + v;
        float a  = acc2[mt][v];
        float tv = tnf[((long long)i * TDIM + t) * EDIM + ecol];
        atomicAdd(&num_lds[t], a * tv);
        atomicAdd(&sq_lds[t], a * a);
      }
  }
  __syncthreads();
  if (tid < TDIM) {
    float l   = l_lds[tid];
    float num = num_lds[tid] / l;
    float w2  = sqrtf(sq_lds[tid]) / l;
    float w1  = w1g[i * TDIM + tid];
    float cosv = num / fmaxf(w1 * w2, 1e-8f);
    red_lds[tid] = __expf(5.0f * cosv);          // TEMP2
    if (diag) lglob[i * TDIM + tid] = l;
  }
  __syncthreads();
  if (tid == 0) {
    float s = 0.f;
    for (int t = 0; t < TDIM; ++t) s += red_lds[t];
    simbuf[b * 16 + i] = 10.0f * logf(s);        // sims[b_vis, i_txt] = row.T * TEMP3
  }
}

// ---------------------------------------------------------------------------
// Kernel 2: att_maps row scale by 1/l[i,t]
// ---------------------------------------------------------------------------
__global__ void att_scale_kernel(float* __restrict__ att, const float* __restrict__ lglob) {
  const int it = blockIdx.x;                     // i*64 + t, 1024 blocks
  const float inv = 1.0f / lglob[it];
  float* p = att + (long long)it * SDIM;
  for (int k = threadIdx.x; k < SDIM; k += blockDim.x) p[k] *= inv;
}

// ---------------------------------------------------------------------------
// Kernel 3: symmetric CE loss over sims[16,16]
// ---------------------------------------------------------------------------
__global__ void loss_kernel(const float* __restrict__ simbuf, float* __restrict__ out) {
  if (threadIdx.x != 0) return;
  float sims[16][16];
  for (int bb = 0; bb < 16; ++bb)
    for (int ii = 0; ii < 16; ++ii) sims[bb][ii] = simbuf[bb * 16 + ii];
  float acc0 = 0.f, acc1 = 0.f;
  for (int bb = 0; bb < 16; ++bb) {              // log_softmax over i (rows)
    float m = -1e30f;
    for (int ii = 0; ii < 16; ++ii) m = fmaxf(m, sims[bb][ii]);
    float s = 0.f;
    for (int ii = 0; ii < 16; ++ii) s += __expf(sims[bb][ii] - m);
    acc0 += sims[bb][bb] - (m + logf(s));
  }
  for (int ii = 0; ii < 16; ++ii) {              // log_softmax over b (cols)
    float m = -1e30f;
    for (int bb = 0; bb < 16; ++bb) m = fmaxf(m, sims[bb][ii]);
    float s = 0.f;
    for (int bb = 0; bb < 16; ++bb) s += __expf(sims[bb][ii] - m);
    acc1 += sims[ii][ii] - (m + logf(s));
  }
  out[0] = -(acc0 / 16.f + acc1 / 16.f) * 0.5f;
}

// ---------------------------------------------------------------------------
extern "C" void kernel_launch(void* const* d_in, const int* in_sizes, int n_in,
                              void* d_out, int out_size, void* d_ws, size_t ws_size,
                              hipStream_t stream) {
  const float* vis = (const float*)d_in[0];      // [16,16,16,16,128] f32
  const float* txt = (const float*)d_in[1];      // [16,64,128] f32
  char* ws = (char*)d_ws;
  __bf16* vnb   = (__bf16*)(ws + 0);             // 16,777,216 B
  __bf16* tnb   = (__bf16*)(ws + 16777216);      //    262,144 B
  float*  tnf   = (float*)(ws + 17039360);       //    524,288 B
  float*  w1    = (float*)(ws + 17563648);       //      4,096 B
  float*  simbf = (float*)(ws + 17567744);       //      1,024 B
  float*  lglob = (float*)(ws + 17568768);       //      4,096 B

  float* out = (float*)d_out;                    // [0]=loss, [1..] att_maps
  float* att = out + 1;

  norm_kernel<<<(BSZ * SDIM + BSZ * TDIM) / 8, 256, 0, stream>>>(vis, txt, vnb, tnb, tnf, w1);
  pair_kernel<<<BSZ * BSZ, 256, 0, stream>>>(vnb, tnb, tnf, w1, simbf, lglob, att);
  att_scale_kernel<<<BSZ * TDIM, 256, 0, stream>>>(att, lglob);
  loss_kernel<<<1, 32, 0, stream>>>(simbf, out);
}
